// RDVCChannelSplitICIPWithGain_28441273434728
// MI455X (gfx1250) — matrix operand order repaired and checked by reading, not compiled, not run on hardware
//
#include <hip/hip_runtime.h>
#include <hip/hip_bf16.h>
#include <math.h>

// ============================================================================
// MI455X (gfx1250, wave32) implementation of the RDVC channel-conditional
// compression forward pass.  All convolutions / transposed convolutions are
// lowered to implicit-GEMM:
//   - B (weight) tiles DMA'd into double-buffered LDS by the Tensor Data
//     Mover, pipelined one K-chunk ahead (tensor_load_to_lds + s_wait_tensorcnt)
//   - A (im2col) tiles staged by VALU with hoisted address math, double-buffered
//   - 4x V_WMMA_F32_16X16X32_F16 per wave per K-chunk (32x32 wave tile)
//   - fragment gathers as ds_load_b128 (2 per fragment)
// ============================================================================

typedef __attribute__((ext_vector_type(16))) _Float16     v16h;
typedef __attribute__((ext_vector_type(8)))  float        v8f;
typedef __attribute__((ext_vector_type(4)))  unsigned int v4u;
typedef __attribute__((ext_vector_type(8)))  int          v8i;
typedef __attribute__((ext_vector_type(4)))  int          v4i;

#define HW16  256          // 16*16 spatial
#define NCH   192
#define SDC   24

#define BM  64
#define BN  64
#define BK  32
#define BKP 40   // padded LDS row stride in f16 (80 bytes): TDM pad 4 dwords / 16 dwords

// ---------------------------------------------------------------------------
// Weight packing: OIHW conv weights (or IOHW transposed-conv weights, which
// get spatially flipped) -> f16 [Npad][Kpad] GEMM-B layout, zero padded.
// ---------------------------------------------------------------------------
__global__ void pack_w_kernel(const float* __restrict__ w, _Float16* __restrict__ wpk,
                              int Cin, int Cout, int ksz, int K, int Kpad, int Npad,
                              int transposed)
{
    int idx = blockIdx.x * blockDim.x + threadIdx.x;
    int tot = Npad * Kpad;
    if (idx >= tot) return;
    int n = idx / Kpad;
    int k = idx - n * Kpad;
    int k2 = ksz * ksz;
    float v = 0.f;
    if (n < Cout && k < K) {
        int cin = k / k2, rem = k - cin * k2;
        if (!transposed) {
            v = w[(n * Cin + cin) * k2 + rem];                      // OIHW
        } else {
            int r = rem / ksz, s = rem - r * ksz;                   // IOHW + flip
            v = w[(cin * Cout + n) * k2 + (ksz - 1 - r) * ksz + (ksz - 1 - s)];
        }
    }
    wpk[idx] = (_Float16)v;
}

// ---- set a bitfield inside a little-endian 256-bit value (4 x u64) --------
__device__ __forceinline__ void dset(unsigned long long* q, int lo, int width,
                                     unsigned long long val)
{
    unsigned long long msk = (width >= 64) ? ~0ull : ((1ull << width) - 1ull);
    val &= msk;
    q[lo >> 6] |= val << (lo & 63);
    int spill = (lo & 63) + width - 64;
    if (spill > 0) q[(lo >> 6) + 1] |= val >> (width - spill);
}

// ---------------------------------------------------------------------------
// Implicit-GEMM conv / lhs-dilated deconv.
//   M = Hout*Wout pixels, N = Cout, K = Cin*k*k
//   block = 128 threads = 4 wave32 waves in 2x2; each wave owns a 32x32 tile
//   dil==1: ordinary conv with `stride`; dil==2: transposed conv (stride==1)
//   act: 0=none, 1=exact GELU, 2=0.5*tanh
// ---------------------------------------------------------------------------
__global__ __launch_bounds__(128) void conv_wmma_kernel(
    const float* __restrict__ in, const _Float16* __restrict__ wpk,
    const float* __restrict__ bias, float* __restrict__ out,
    int Hin, int Win, int lw, int wmask, int Cout,
    int K, int Kpad, int Npad, int ksz, int stride, int dil, int pad, int act, int M)
{
    extern __shared__ __align__(16) _Float16 smem[];
    // LDS layout: As[0] | As[1] | Bs[0] | Bs[1]; compute pointers on the fly
    // (no pointer aggregates: LDS addrspacecast in static initializers breaks lld)
    auto Abuf = [&](int p) -> _Float16* { return smem + p * (BM * BKP); };
    auto Bbuf = [&](int p) -> _Float16* { return smem + 2 * BM * BKP + p * (BN * BKP); };
    const int ldsB0      = 2 * BM * BKP * (int)sizeof(_Float16);
    const int ldsBstride = BN * BKP * (int)sizeof(_Float16);

    const int tid  = threadIdx.x;
    const int lane = tid & 31;
    const int wave = tid >> 5;
    const int wm   = wave >> 1;                // wave row (M)
    const int wn   = wave & 1;                 // wave col (N)
    const int half = lane >> 4;
    const int l16  = lane & 15;
    const int k2   = ksz * ksz;
    const int m0   = blockIdx.x * BM;
    const int n0   = blockIdx.y * BN;

    v8f acc0 = {}, acc1 = {}, acc2 = {}, acc3 = {};

    const int nk = Kpad >> 5;
    const int c  = tid & 31;                   // K column: invariant per thread
    const int r0 = tid >> 5;

    // ---- issue the TDM DMA of a 64x32 f16 weight tile into LDS buffer p ----
    auto issue_tdm = [&](int k0, int p) {
        unsigned long long gaddr =
            (unsigned long long)(const void*)(wpk + (size_t)n0 * Kpad + k0);
        v4u g0;
        g0[0] = 1u;                                          // count=1
        g0[1] = (unsigned)(ldsB0 + p * ldsBstride);          // lds_addr
        g0[2] = (unsigned)gaddr;                             // global_addr lo
        g0[3] = (unsigned)((gaddr >> 32) & 0x01ffffffull)    // global_addr hi
                | 0x80000000u;                               // type=2 (image)
        unsigned long long q[4] = {0ull, 0ull, 0ull, 0ull};
        dset(q,  16,  2, 1);                 // data_size = 2 bytes
        dset(q,  20,  1, 1);                 // pad_enable
        dset(q,  22,  3, 3);                 // pad interval: 16 DWORDs (64B row)
        dset(q,  25,  7, 3);                 // pad amount: 4 DWORDs -> 80B stride
        dset(q,  48, 32, (unsigned)Kpad);    // tensor_dim0
        dset(q,  80, 32, (unsigned)Npad);    // tensor_dim1
        dset(q, 112, 16, BK);                // tile_dim0 (K)
        dset(q, 128, 16, BN);                // tile_dim1 (rows)
        dset(q, 160, 48, (unsigned)Kpad);    // tensor_dim0_stride
        v8i g1;
        #pragma unroll
        for (int t = 0; t < 4; ++t) {
            g1[2 * t]     = (int)(q[t] & 0xffffffffull);
            g1[2 * t + 1] = (int)(q[t] >> 32);
        }
        v4i gz = {0, 0, 0, 0};
#if defined(__clang_major__) && (__clang_major__ >= 23)
        v8i gz8 = {0, 0, 0, 0, 0, 0, 0, 0};
        __builtin_amdgcn_tensor_load_to_lds(g0, g1, gz, gz, gz8, 0);
#else
        __builtin_amdgcn_tensor_load_to_lds(g0, g1, gz, gz, 0);
#endif
    };

    // ---- stage a 64x32 im2col A tile; K-decomposition hoisted per chunk ----
    auto stage_a = [&](int k0, int p) {
        _Float16* As = Abuf(p);
        const int kk = k0 + c;
        int cin = 0, rr = 0, ss = 0;
        const bool kval = (kk < K);
        if (kval) {
            cin = kk / k2; int rem = kk - cin * k2;
            rr = rem / ksz; ss = rem - rr * ksz;
        }
        const float* inc = in + (size_t)cin * Hin * Win;
        #pragma unroll
        for (int it = 0; it < 16; ++it) {
            int r = r0 + it * 4;
            int m = m0 + r;
            float va = 0.f;
            if (kval && m < M) {
                int oh  = m >> lw, ow = m & wmask;
                int ihr = oh * stride - pad + rr;
                int iwr = ow * stride - pad + ss;
                if (ihr >= 0 && iwr >= 0) {
                    if (dil == 1) {
                        if (ihr < Hin && iwr < Win)
                            va = inc[ihr * Win + iwr];
                    } else {  // lhs_dilation == 2 (transposed conv)
                        if (!(ihr & 1) && !(iwr & 1)) {
                            int ih = ihr >> 1, iw = iwr >> 1;
                            if (ih < Hin && iw < Win)
                                va = inc[ih * Win + iw];
                        }
                    }
                }
            }
            As[r * BKP + c] = (_Float16)va;
        }
    };

    // ---- prologue: fill buffer 0 ----
    if (wave == 0) issue_tdm(0, 0);
    stage_a(0, 0);

    for (int kt = 0; kt < nk; ++kt) {
        const int p     = kt & 1;
        const bool more = (kt + 1 < nk);

        if (wave == 0) __builtin_amdgcn_s_wait_tensorcnt(0);  // TDM(kt) complete
        __syncthreads();   // A(kt) visible; all waves done with buffer 1-p

        // pipeline: DMA the next weight tile into the idle buffer now
        if (more && wave == 0) issue_tdm((kt + 1) << 5, 1 - p);
        if (more && wave == 1)  // warm GL2 two chunks ahead of the TDM pulls
            __builtin_prefetch(wpk + (size_t)(n0 + lane) * Kpad + ((kt + 2) << 5), 0, 1);

        // ---- fragments (gfx1250 16-bit A/B VGPR layouts), b128 LDS loads ----
        union FragU { v4u q[2]; unsigned int u[8]; v16h h; };
        FragU a0, a1, b0, b1;
        const _Float16* ar0 = Abuf(p) + (32 * wm + l16) * BKP;
        const _Float16* ar1 = ar0 + 16 * BKP;
        const _Float16* br0 = Bbuf(p) + (32 * wn + l16) * BKP;
        const _Float16* br1 = br0 + 16 * BKP;
        a0.q[0] = *(const v4u*)(ar0 + 8 * half);
        a0.q[1] = *(const v4u*)(ar0 + 16 + 8 * half);
        a1.q[0] = *(const v4u*)(ar1 + 8 * half);
        a1.q[1] = *(const v4u*)(ar1 + 16 + 8 * half);
        b0.q[0] = *(const v4u*)(br0 + 8 * half);
        b0.q[1] = *(const v4u*)(br0 + 16 + 8 * half);
        b1.q[0] = *(const v4u*)(br1 + 8 * half);
        b1.q[1] = *(const v4u*)(br1 + 16 + 8 * half);

        acc0 = __builtin_amdgcn_wmma_f32_16x16x32_f16(false, a0.h, false, b0.h, (short)0, acc0, false, false);
        acc1 = __builtin_amdgcn_wmma_f32_16x16x32_f16(false, a0.h, false, b1.h, (short)0, acc1, false, false);
        acc2 = __builtin_amdgcn_wmma_f32_16x16x32_f16(false, a1.h, false, b0.h, (short)0, acc2, false, false);
        acc3 = __builtin_amdgcn_wmma_f32_16x16x32_f16(false, a1.h, false, b1.h, (short)0, acc3, false, false);

        // stage the next A tile into the idle buffer (overlaps the TDM DMA)
        if (more) stage_a((kt + 1) << 5, 1 - p);
    }

    // ---- epilogue: bias + activation, write NCHW fp32 ----
    const int nb = n0 + 32 * wn;
    const int mb = m0 + 32 * wm;
    #pragma unroll
    for (int idx = 0; idx < 4; ++idx) {
        const int i2 = idx >> 1;            // M sub-tile
        const int j  = idx & 1;             // N sub-tile
        v8f a = (idx == 0) ? acc0 : (idx == 1) ? acc1 : (idx == 2) ? acc2 : acc3;
        int n = nb + 16 * j + l16;
        if (n < Cout) {
            float bv = bias[n];
            #pragma unroll
            for (int r = 0; r < 8; ++r) {
                int m = mb + 16 * i2 + 8 * half + r;
                if (m < M) {
                    float v = a[r] + bv;
                    if (act == 1)      v = 0.5f * v * erfcf(-v * 0.70710678118654752f);
                    else if (act == 2) v = 0.5f * tanhf(v);
                    out[n * M + m] = v;
                }
            }
        }
    }
}

// ---------------------------------------------------------------------------
// Elementwise kernels
// ---------------------------------------------------------------------------
__global__ void ew_z_kernel(const float* __restrict__ z, const float* __restrict__ med,
                            float* __restrict__ zlik, float* __restrict__ zhat,
                            int HW, int n)
{
    int i = blockIdx.x * blockDim.x + threadIdx.x;
    if (i >= n) return;
    float m  = med[i / HW];
    float zc = z[i] - m;
    float s1 = 1.f / (1.f + expf(-(zc + 0.5f)));
    float s0 = 1.f / (1.f + expf(-(zc - 0.5f)));
    zlik[i] = fmaxf(s1 - s0, 1e-9f);
    zhat[i] = rintf(zc) + m;          // round-half-even, matches jnp.round
}

__global__ void ew_qs_kernel(const float* __restrict__ q, float* __restrict__ qs, int n)
{
    int i = blockIdx.x * blockDim.x + threadIdx.x;
    if (i < n) qs[i] = fmaxf(q[i], 0.5f);
}

__global__ void ew_div_kernel(const float* __restrict__ a, const float* __restrict__ b,
                              float* __restrict__ o, int n)
{
    int i = blockIdx.x * blockDim.x + threadIdx.x;
    if (i < n) o[i] = a[i] / b[i];
}

__global__ void ew_lik_kernel(const float* __restrict__ ys, const float* __restrict__ mu,
                              const float* __restrict__ sc, float* __restrict__ lik, int n)
{
    int i = blockIdx.x * blockDim.x + threadIdx.x;
    if (i >= n) return;
    float s  = fmaxf(sc[i], 0.11f);
    float v  = fabsf(ys[i] - mu[i]);
    float c1 = 0.5f * erfcf(-((0.5f - v) / s) * 0.70710678118654752f);
    float c0 = 0.5f * erfcf(-((-0.5f - v) / s) * 0.70710678118654752f);
    lik[i] = fmaxf(c1 - c0, 1e-9f);
}

__global__ void ew_yh_kernel(const float* __restrict__ ys, const float* __restrict__ mu,
                             float* __restrict__ yh, int n)
{
    int i = blockIdx.x * blockDim.x + threadIdx.x;
    if (i < n) yh[i] = rintf(ys[i] - mu[i]) + mu[i];
}

__global__ void ew_add_kernel(float* __restrict__ a, const float* __restrict__ b, int n)
{
    int i = blockIdx.x * blockDim.x + threadIdx.x;
    if (i < n) a[i] += b[i];
}

__global__ void ew_mul_kernel(float* __restrict__ a, const float* __restrict__ b, int n)
{
    int i = blockIdx.x * blockDim.x + threadIdx.x;
    if (i < n) a[i] *= b[i];
}

// ---------------------------------------------------------------------------
// Host-side: parameter index mapping (supports both insertion-order and
// jax-sorted-key flattening of the setup_inputs() pytree).
// ---------------------------------------------------------------------------
struct ParamIdx {
    int ga_w[4], ga_b[4];
    int ha_w[3], ha_b[3];
    int hs_w[3], hs_b[3];
    int gs_w[4], gs_b[4];
    int fea_w[2], fea_b[2];
    int par_w[2], par_b[2];
    int ccm_w[8][4], ccm_b[8][4];
    int ccs_w[8][4], ccs_b[8][4];
    int lrp_w[8][3], lrp_b[8][3];
    int medians, x, feature;
};

static void build_idx(ParamIdx& I, bool sorted)
{
    int p = 0;
    if (!sorted) {  // insertion order: params{g_a,h_a,h_s,g_s,fea,param,cc_mean,cc_scale,lrp,medians}, x, feature; each conv {w,b}
        for (int i = 0; i < 4; ++i) { I.ga_w[i] = p++; I.ga_b[i] = p++; }
        for (int i = 0; i < 3; ++i) { I.ha_w[i] = p++; I.ha_b[i] = p++; }
        for (int i = 0; i < 3; ++i) { I.hs_w[i] = p++; I.hs_b[i] = p++; }
        for (int i = 0; i < 4; ++i) { I.gs_w[i] = p++; I.gs_b[i] = p++; }
        for (int i = 0; i < 2; ++i) { I.fea_w[i] = p++; I.fea_b[i] = p++; }
        for (int i = 0; i < 2; ++i) { I.par_w[i] = p++; I.par_b[i] = p++; }
        for (int i = 0; i < 8; ++i) for (int j = 0; j < 4; ++j) { I.ccm_w[i][j] = p++; I.ccm_b[i][j] = p++; }
        for (int i = 0; i < 8; ++i) for (int j = 0; j < 4; ++j) { I.ccs_w[i][j] = p++; I.ccs_b[i][j] = p++; }
        for (int i = 0; i < 8; ++i) for (int j = 0; j < 3; ++j) { I.lrp_w[i][j] = p++; I.lrp_b[i][j] = p++; }
        I.medians = p++; I.x = p++; I.feature = p++;
    } else {        // jax tree_flatten: sorted keys; 'b' < 'w'
        I.feature = p++;
        for (int i = 0; i < 8; ++i) for (int j = 0; j < 4; ++j) { I.ccm_b[i][j] = p++; I.ccm_w[i][j] = p++; }
        for (int i = 0; i < 8; ++i) for (int j = 0; j < 4; ++j) { I.ccs_b[i][j] = p++; I.ccs_w[i][j] = p++; }
        for (int i = 0; i < 2; ++i) { I.fea_b[i] = p++; I.fea_w[i] = p++; }
        for (int i = 0; i < 4; ++i) { I.ga_b[i]  = p++; I.ga_w[i]  = p++; }
        for (int i = 0; i < 4; ++i) { I.gs_b[i]  = p++; I.gs_w[i]  = p++; }
        for (int i = 0; i < 3; ++i) { I.ha_b[i]  = p++; I.ha_w[i]  = p++; }
        for (int i = 0; i < 3; ++i) { I.hs_b[i]  = p++; I.hs_w[i]  = p++; }
        for (int i = 0; i < 8; ++i) for (int j = 0; j < 3; ++j) { I.lrp_b[i][j] = p++; I.lrp_w[i][j] = p++; }
        I.medians = p++;
        for (int i = 0; i < 2; ++i) { I.par_b[i] = p++; I.par_w[i] = p++; }
        I.x = p++;
    }
}

static void conv_layer(hipStream_t st, void* wpk,
                       const float* in, const float* w, const float* b, float* out,
                       int Cin, int Hin, int Win, int Cout, int Hout, int Wout,
                       int ksz, int stride, int dil, int pad, int transposed, int act)
{
    int k2   = ksz * ksz;
    int K    = Cin * k2;
    int Kpad = (K + 31) & ~31;
    int Npad = (Cout + 63) & ~63;
    int tot  = Npad * Kpad;
    pack_w_kernel<<<(tot + 255) / 256, 256, 0, st>>>(
        w, (_Float16*)wpk, Cin, Cout, ksz, K, Kpad, Npad, transposed);
    int M  = Hout * Wout;
    int lw = 0; while ((1 << lw) < Wout) ++lw;
    dim3 grid((M + BM - 1) / BM, Npad / BN);
    size_t smem = (size_t)2 * (BM + BN) * BKP * sizeof(_Float16);   // double-buffered
    conv_wmma_kernel<<<grid, 128, smem, st>>>(
        in, (const _Float16*)wpk, b, out,
        Hin, Win, lw, Wout - 1, Cout, K, Kpad, Npad, ksz, stride, dil, pad, act, M);
}

// ---------------------------------------------------------------------------
extern "C" void kernel_launch(void* const* d_in, const int* in_sizes, int n_in,
                              void* d_out, int out_size, void* d_ws, size_t ws_size,
                              hipStream_t stream)
{
    (void)out_size; (void)ws_size;
    ParamIdx I;
    bool sorted = (n_in > 0 && in_sizes[0] == 16384);   // feature-first => sorted keys
    build_idx(I, sorted);
    auto F = [&](int idx) -> const float* { return (const float*)d_in[idx]; };

    // ---- workspace partition ----
    char* cur = (char*)d_ws;
    auto alloc = [&](size_t bytes) -> char* {
        char* p = cur; cur += (bytes + 255) & ~(size_t)255; return p;
    };
    float* bufA  = (float*)alloc(192ull * 128 * 128 * 4);
    float* bufB  = (float*)alloc(192ull * 128 * 128 * 4);
    float* y     = (float*)alloc(NCH * HW16 * 4);
    float* z     = (float*)alloc(NCH * 16 * 4);
    float* zhat  = (float*)alloc(NCH * 16 * 4);
    float* t1    = (float*)alloc(NCH * HW16 * 4);
    float* t2    = (float*)alloc(NCH * HW16 * 4);
    float* hf    = (float*)alloc(2 * NCH * HW16 * 4);   // concat(h, f)
    float* lat   = (float*)alloc(3 * NCH * HW16 * 4);   // q | scales | means
    float* qs    = (float*)alloc(NCH * HW16 * 4);
    float* ynorm = (float*)alloc(NCH * HW16 * 4);
    float* yhat  = (float*)alloc(NCH * HW16 * 4);
    float* msup  = (float*)alloc((NCH + 4 * SDC) * HW16 * 4);
    float* ssup  = (float*)alloc((NCH + 4 * SDC) * HW16 * 4);
    float* lrpin = (float*)alloc((NCH + 5 * SDC) * HW16 * 4);
    float* mu    = (float*)alloc(SDC * HW16 * 4);
    float* sc    = (float*)alloc(SDC * HW16 * 4);
    float* lrpv  = (float*)alloc(SDC * HW16 * 4);
    void*  wpk   = alloc(4ull * 1024 * 1024);           // max packed weight (f16)

    const float* x   = F(I.x);
    const float* fea = F(I.feature);
    const float* med = F(I.medians);
    float* outF  = (float*)d_out;                 // xh: 3x256x256
    float* ylikO = outF + 3 * 256 * 256;          // y_liks: 192x16x16
    float* zlikO = ylikO + NCH * HW16;            // z_lik: 192x4x4

    // ---- g_a: 256^2 -> 16^2 ----
    conv_layer(stream, wpk, x,    F(I.ga_w[0]), F(I.ga_b[0]), bufA,   3, 256, 256, NCH, 128, 128, 5, 2, 1, 2, 0, 1);
    conv_layer(stream, wpk, bufA, F(I.ga_w[1]), F(I.ga_b[1]), bufB, NCH, 128, 128, NCH,  64,  64, 5, 2, 1, 2, 0, 1);
    conv_layer(stream, wpk, bufB, F(I.ga_w[2]), F(I.ga_b[2]), bufA, NCH,  64,  64, NCH,  32,  32, 5, 2, 1, 2, 0, 1);
    conv_layer(stream, wpk, bufA, F(I.ga_w[3]), F(I.ga_b[3]), y,    NCH,  32,  32, NCH,  16,  16, 5, 2, 1, 2, 0, 0);

    // ---- h_a: 16^2 -> 4^2 ----
    conv_layer(stream, wpk, y,  F(I.ha_w[0]), F(I.ha_b[0]), t1, NCH, 16, 16, NCH, 16, 16, 3, 1, 1, 1, 0, 1);
    conv_layer(stream, wpk, t1, F(I.ha_w[1]), F(I.ha_b[1]), t2, NCH, 16, 16, NCH,  8,  8, 5, 2, 1, 2, 0, 1);
    conv_layer(stream, wpk, t2, F(I.ha_w[2]), F(I.ha_b[2]), z,  NCH,  8,  8, NCH,  4,  4, 5, 2, 1, 2, 0, 0);

    // ---- z likelihoods + z_hat ----
    ew_z_kernel<<<(NCH * 16 + 255) / 256, 256, 0, stream>>>(z, med, zlikO, zhat, 16, NCH * 16);

    // ---- fea: feature(64ch) -> f (second half of hf) ----
    conv_layer(stream, wpk, fea, F(I.fea_w[0]), F(I.fea_b[0]), t1,              64, 16, 16, NCH, 16, 16, 3, 1, 1, 1, 0, 1);
    conv_layer(stream, wpk, t1,  F(I.fea_w[1]), F(I.fea_b[1]), hf + NCH * HW16, NCH, 16, 16, NCH, 16, 16, 3, 1, 1, 1, 0, 0);

    // ---- h_s: 4^2 -> 16^2, h into first half of hf ----
    conv_layer(stream, wpk, zhat, F(I.hs_w[0]), F(I.hs_b[0]), t1, NCH,  4,  4, NCH,  8,  8, 5, 1, 2, 2, 1, 1);
    conv_layer(stream, wpk, t1,   F(I.hs_w[1]), F(I.hs_b[1]), t2, NCH,  8,  8, NCH, 16, 16, 5, 1, 2, 2, 1, 1);
    conv_layer(stream, wpk, t2,   F(I.hs_w[2]), F(I.hs_b[2]), hf, NCH, 16, 16, NCH, 16, 16, 3, 1, 1, 1, 0, 0);

    // ---- param head: concat(h,f)(384) -> 576 = q | scales | means ----
    conv_layer(stream, wpk, hf, F(I.par_w[0]), F(I.par_b[0]), t1,  2 * NCH, 16, 16, NCH,     16, 16, 3, 1, 1, 1, 0, 1);
    conv_layer(stream, wpk, t1, F(I.par_w[1]), F(I.par_b[1]), lat, NCH,     16, 16, 3 * NCH, 16, 16, 3, 1, 1, 1, 0, 0);

    int nY = NCH * HW16;
    ew_qs_kernel <<<(nY + 255) / 256, 256, 0, stream>>>(lat, qs, nY);
    ew_div_kernel<<<(nY + 255) / 256, 256, 0, stream>>>(y, qs, ynorm, nY);

    // support buffers seeded with latent means / scales (channel-contiguous concat)
    (void)hipMemcpyAsync(msup, lat + 2 * NCH * HW16, (size_t)nY * 4, hipMemcpyDeviceToDevice, stream);
    (void)hipMemcpyAsync(ssup, lat + 1 * NCH * HW16, (size_t)nY * 4, hipMemcpyDeviceToDevice, stream);

    int nS = SDC * HW16;
    for (int i = 0; i < 8; ++i) {
        int ms   = (i < 4) ? i : 4;
        int cinM = NCH + SDC * ms;
        // cc_mean: cinM->192->64->32->24
        conv_layer(stream, wpk, msup, F(I.ccm_w[i][0]), F(I.ccm_b[i][0]), t1, cinM, 16, 16, NCH, 16, 16, 3, 1, 1, 1, 0, 1);
        conv_layer(stream, wpk, t1,   F(I.ccm_w[i][1]), F(I.ccm_b[i][1]), t2, NCH,  16, 16,  64, 16, 16, 3, 1, 1, 1, 0, 1);
        conv_layer(stream, wpk, t2,   F(I.ccm_w[i][2]), F(I.ccm_b[i][2]), t1,  64,  16, 16,  32, 16, 16, 3, 1, 1, 1, 0, 1);
        conv_layer(stream, wpk, t1,   F(I.ccm_w[i][3]), F(I.ccm_b[i][3]), mu,  32,  16, 16, SDC, 16, 16, 3, 1, 1, 1, 0, 0);
        // cc_scale
        conv_layer(stream, wpk, ssup, F(I.ccs_w[i][0]), F(I.ccs_b[i][0]), t1, cinM, 16, 16, NCH, 16, 16, 3, 1, 1, 1, 0, 1);
        conv_layer(stream, wpk, t1,   F(I.ccs_w[i][1]), F(I.ccs_b[i][1]), t2, NCH,  16, 16,  64, 16, 16, 3, 1, 1, 1, 0, 1);
        conv_layer(stream, wpk, t2,   F(I.ccs_w[i][2]), F(I.ccs_b[i][2]), t1,  64,  16, 16,  32, 16, 16, 3, 1, 1, 1, 0, 1);
        conv_layer(stream, wpk, t1,   F(I.ccs_w[i][3]), F(I.ccs_b[i][3]), sc,  32,  16, 16, SDC, 16, 16, 3, 1, 1, 1, 0, 0);

        const float* ysl = ynorm + (size_t)i * nS;
        float*       yhi = yhat  + (size_t)i * nS;
        ew_lik_kernel<<<(nS + 255) / 256, 256, 0, stream>>>(ysl, mu, sc, ylikO + (size_t)i * nS, nS);
        ew_yh_kernel <<<(nS + 255) / 256, 256, 0, stream>>>(ysl, mu, yhi, nS);

        // lrp input = concat(mean_support, yh)
        (void)hipMemcpyAsync(lrpin, msup, (size_t)cinM * HW16 * 4, hipMemcpyDeviceToDevice, stream);
        (void)hipMemcpyAsync(lrpin + (size_t)cinM * HW16, yhi, (size_t)nS * 4, hipMemcpyDeviceToDevice, stream);
        int cinL = cinM + SDC;    // == 192 + 24*min(i+1,5)
        conv_layer(stream, wpk, lrpin, F(I.lrp_w[i][0]), F(I.lrp_b[i][0]), t1,   cinL,    16, 16, NCH,     16, 16, 3, 1, 1, 1, 0, 1);
        conv_layer(stream, wpk, t1,    F(I.lrp_w[i][1]), F(I.lrp_b[i][1]), t2,   NCH,     16, 16, NCH / 2, 16, 16, 3, 1, 1, 1, 0, 1);
        conv_layer(stream, wpk, t2,    F(I.lrp_w[i][2]), F(I.lrp_b[i][2]), lrpv, NCH / 2, 16, 16, SDC,     16, 16, 3, 1, 1, 1, 0, 2);
        ew_add_kernel<<<(nS + 255) / 256, 256, 0, stream>>>(yhi, lrpv, nS);

        if (i < 4) {  // append finalized slice to both supports
            (void)hipMemcpyAsync(msup + (size_t)(NCH + SDC * i) * HW16, yhi, (size_t)nS * 4, hipMemcpyDeviceToDevice, stream);
            (void)hipMemcpyAsync(ssup + (size_t)(NCH + SDC * i) * HW16, yhi, (size_t)nS * 4, hipMemcpyDeviceToDevice, stream);
        }
    }

    ew_mul_kernel<<<(nY + 255) / 256, 256, 0, stream>>>(yhat, qs, nY);

    // ---- g_s: 16^2 -> 256^2, final layer straight into d_out ----
    conv_layer(stream, wpk, yhat, F(I.gs_w[0]), F(I.gs_b[0]), bufA, NCH, 16,  16,  NCH, 32,  32,  5, 1, 2, 2, 1, 1);
    conv_layer(stream, wpk, bufA, F(I.gs_w[1]), F(I.gs_b[1]), bufB, NCH, 32,  32,  NCH, 64,  64,  5, 1, 2, 2, 1, 1);
    conv_layer(stream, wpk, bufB, F(I.gs_w[2]), F(I.gs_b[2]), bufA, NCH, 64,  64,  NCH, 128, 128, 5, 1, 2, 2, 1, 1);
    conv_layer(stream, wpk, bufA, F(I.gs_w[3]), F(I.gs_b[3]), outF, NCH, 128, 128, 3,   256, 256, 5, 1, 2, 2, 1, 0);
}
